// ConvSelfAttention_31413390803450
// MI455X (gfx1250) — matrix operand order repaired and checked by reading, not compiled, not run on hardware
//
#include <hip/hip_runtime.h>

typedef __attribute__((ext_vector_type(16))) __bf16 v16bf;
typedef __attribute__((ext_vector_type(8)))  __bf16 v8bf;
typedef __attribute__((ext_vector_type(8)))  float  v8f;
typedef __attribute__((ext_vector_type(4)))  float  v4f;

constexpr int kB = 4;
constexpr int kS = 2048;
constexpr int kH = 16;
constexpr int kD = 64;          // head dim
constexpr int kModel = 1024;    // kH * kD
constexpr int kKeysPerIter = 64;
constexpr int kWaves = 8;
constexpr int kQPerWave = 16;
constexpr int kQPerBlock = kWaves * kQPerWave;   // 128
constexpr int kIters = kS / kKeysPerIter;        // 32

// LDS row strides in bf16 units (padded: +8 halves = 16B, keeps 16B alignment,
// makes row stride 144B = 36 dwords -> 16 distinct banks across 16 lanes)
constexpr int KROW  = 72;  // sK : [key][dim], 64 dims + 8 pad
constexpr int KTROW = 72;  // sKT: [dim][key], 64 keys + 8 pad
constexpr int PROW  = 72;  // sP : [qrow][key], 64 keys + 8 pad

__device__ __forceinline__ v8f wmma_bf16(v16bf a, v16bf b, v8f c) {
  // D = A(16x32 bf16) * B(32x16 bf16) + C(16x16 f32)
  return __builtin_amdgcn_wmma_f32_16x16x32_bf16(false, a, false, b,
                                                 (short)0, c, false, false);
}

__device__ __forceinline__ v16bf cat8(v8bf lo, v8bf hi) {
  return __builtin_shufflevector(lo, hi, 0, 1, 2, 3, 4, 5, 6, 7,
                                 8, 9, 10, 11, 12, 13, 14, 15);
}

// ---- xor-butterfly within each 16-lane half, fused shuffle+combine ----
// xor1/xor2 are quad_perm DPP, xor8 is row_ror:8 (rotate by half-row == xor 8),
// xor4 uses v_permlane16_b32 (EXEC is all ones -> tied 'old' is irrelevant).

__device__ __forceinline__ float maxx1(float x) {
  float d;
  asm("v_max_num_f32_dpp %0, %1, %1 quad_perm:[1,0,3,2] row_mask:0xf bank_mask:0xf"
      : "=&v"(d) : "v"(x));
  return d;
}
__device__ __forceinline__ float maxx2(float x) {
  float d;
  asm("v_max_num_f32_dpp %0, %1, %1 quad_perm:[2,3,0,1] row_mask:0xf bank_mask:0xf"
      : "=&v"(d) : "v"(x));
  return d;
}
__device__ __forceinline__ float maxx8(float x) {
  float d;
  asm("v_max_num_f32_dpp %0, %1, %1 row_ror:8 row_mask:0xf bank_mask:0xf"
      : "=&v"(d) : "v"(x));
  return d;
}
__device__ __forceinline__ float maxx4(float x) {
  float d;
  asm("v_permlane16_b32 %0, %1, %2, %3\n\t"
      "v_max_num_f32_e32 %0, %1, %0"
      : "=&v"(d) : "v"(x), "s"(0x32107654u), "s"(0xBA98FEDCu));
  return d;
}

__device__ __forceinline__ float addx1(float x) {
  float d;
  asm("v_add_f32_dpp %0, %1, %1 quad_perm:[1,0,3,2] row_mask:0xf bank_mask:0xf"
      : "=&v"(d) : "v"(x));
  return d;
}
__device__ __forceinline__ float addx2(float x) {
  float d;
  asm("v_add_f32_dpp %0, %1, %1 quad_perm:[2,3,0,1] row_mask:0xf bank_mask:0xf"
      : "=&v"(d) : "v"(x));
  return d;
}
__device__ __forceinline__ float addx8(float x) {
  float d;
  asm("v_add_f32_dpp %0, %1, %1 row_ror:8 row_mask:0xf bank_mask:0xf"
      : "=&v"(d) : "v"(x));
  return d;
}
__device__ __forceinline__ float addx4(float x) {
  float d;
  asm("v_permlane16_b32 %0, %1, %2, %3\n\t"
      "v_add_f32_e32 %0, %1, %0"
      : "=&v"(d) : "v"(x), "s"(0x32107654u), "s"(0xBA98FEDCu));
  return d;
}

__device__ __forceinline__ float redmax16(float t) {
  t = maxx1(t);
  t = maxx2(t);
  t = maxx4(t);
  t = maxx8(t);
  return t;
}
__device__ __forceinline__ float redsum16(float t) {
  t = addx1(t);
  t = addx2(t);
  t = addx4(t);
  t = addx8(t);
  return t;
}

__global__ __launch_bounds__(256)
void ConvSelfAttention_fa_kernel(const float* __restrict__ Q,
                                 float* __restrict__ Out) {
  __shared__ __bf16 sK [2][kKeysPerIter * KROW];   // row-major keys
  __shared__ __bf16 sKT[2][kD * KTROW];            // transposed keys (= V^T feed)
  __shared__ __bf16 sP [kWaves][kQPerWave * PROW]; // per-wave P transpose scratch

  const int tid  = threadIdx.x;
  const int lane = tid & 31;
  const int wave = tid >> 5;
  const int m    = lane & 15;
  const int hi   = lane >> 4;

  const int h = blockIdx.y;
  const int b = blockIdx.z;
  const float* Qh = Q + ((size_t)b * kS) * kModel + h * kD;
  const int q0 = blockIdx.x * kQPerBlock + wave * kQPerWave;

  // ---- Q tile -> two A fragments, pre-scaled by log2(e)/sqrt(64) ----
  const float scale = 0.125f * 1.44269504088896340736f;  // exp2 domain
  v16bf aQ[2];
  {
    const float* qrow = Qh + (size_t)(q0 + m) * kModel;
    #pragma unroll
    for (int c = 0; c < 2; ++c) {
      v8bf lo, hh;
      const int d0 = 32 * c + hi * 16;
      v4f x0 = *(const v4f*)(qrow + d0);
      v4f x1 = *(const v4f*)(qrow + d0 + 4);
      const int d1 = 32 * c + hi * 16 + 8;
      v4f y0 = *(const v4f*)(qrow + d1);
      v4f y1 = *(const v4f*)(qrow + d1 + 4);
      #pragma unroll
      for (int i = 0; i < 4; ++i) {
        lo[i]     = (__bf16)(x0[i] * scale);
        lo[4 + i] = (__bf16)(x1[i] * scale);
        hh[i]     = (__bf16)(y0[i] * scale);
        hh[4 + i] = (__bf16)(y1[i] * scale);
      }
      aQ[c] = cat8(lo, hh);
    }
  }

  v8f acc[4] = {};
  float mrow[8], lrow[8];
  #pragma unroll
  for (int r = 0; r < 8; ++r) { mrow[r] = -__builtin_inff(); lrow[r] = 0.0f; }

  // ---- cooperative staging: thread t -> key = t>>2, 16 consecutive dims ----
  const int skey  = tid >> 2;          // 0..63
  const int sdcol = (tid & 3) * 16;    // 0,16,32,48
  v4f sx0, sx1, sx2, sx3;
  {
    const float* p = Qh + (size_t)skey * kModel + sdcol;
    sx0 = *(const v4f*)p;       sx1 = *(const v4f*)(p + 4);
    sx2 = *(const v4f*)(p + 8); sx3 = *(const v4f*)(p + 12);
  }

  for (int it = 0; it < kIters; ++it) {
    const int buf = it & 1;

    // ---- stage regs -> LDS (bf16, row-major + transposed) ----
    {
      v8bf h0, h1;
      #pragma unroll
      for (int i = 0; i < 4; ++i) {
        h0[i]     = (__bf16)sx0[i];
        h0[4 + i] = (__bf16)sx1[i];
        h1[i]     = (__bf16)sx2[i];
        h1[4 + i] = (__bf16)sx3[i];
      }
      *(v8bf*)(&sK[buf][skey * KROW + sdcol])     = h0;
      *(v8bf*)(&sK[buf][skey * KROW + sdcol + 8]) = h1;
      #pragma unroll
      for (int i = 0; i < 8; ++i) {
        sKT[buf][(sdcol + i)     * KTROW + skey] = h0[i];
        sKT[buf][(sdcol + 8 + i) * KTROW + skey] = h1[i];
      }
    }
    __syncthreads();

    // prefetch key stream: regs for it+1, cache prefetch for it+2
    if (it + 1 < kIters) {
      const float* p = Qh + (size_t)((it + 1) * kKeysPerIter + skey) * kModel + sdcol;
      sx0 = *(const v4f*)p;       sx1 = *(const v4f*)(p + 4);
      sx2 = *(const v4f*)(p + 8); sx3 = *(const v4f*)(p + 12);
    }
    if (it + 2 < kIters) {
      const float* pf = Qh + (size_t)((it + 2) * kKeysPerIter + skey) * kModel + sdcol;
      __builtin_prefetch(pf, 0, 0);   // -> global_prefetch_b8
    }

    // ---- scores: S[16q x 64k] via 8 WMMAs ----
    const __bf16* kb = sK[buf];
    v8f sc[4];
    #pragma unroll
    for (int jt = 0; jt < 4; ++jt) {
      const __bf16* p = kb + (jt * 16 + m) * KROW + hi * 16;
      v16bf bk0 = cat8(*(const v8bf*)p,        *(const v8bf*)(p + 8));
      v16bf bk1 = cat8(*(const v8bf*)(p + 32), *(const v8bf*)(p + 40));
      v8f s = {};
      s = wmma_bf16(aQ[0], bk0, s);
      s = wmma_bf16(aQ[1], bk1, s);
      sc[jt] = s;
    }

    // ---- online softmax in exp2 domain ----
    float corr[8];
    #pragma unroll
    for (int r = 0; r < 8; ++r) {
      float t = fmaxf(fmaxf(sc[0][r], sc[1][r]), fmaxf(sc[2][r], sc[3][r]));
      t = redmax16(t);
      const float mn = fmaxf(mrow[r], t);
      const float cr = __builtin_amdgcn_exp2f(mrow[r] - mn);
      mrow[r] = mn;
      corr[r] = cr;
      float su = 0.0f;
      #pragma unroll
      for (int jt = 0; jt < 4; ++jt) {
        const float pj = __builtin_amdgcn_exp2f(sc[jt][r] - mn);
        sc[jt][r] = pj;
        su += pj;
      }
      su = redsum16(su);
      lrow[r] = lrow[r] * cr + su;
    }
    #pragma unroll
    for (int dt = 0; dt < 4; ++dt)
      #pragma unroll
      for (int r = 0; r < 8; ++r) acc[dt][r] *= corr[r];

    // ---- transpose P (16x64) through per-wave LDS, reload as A fragments ----
    __bf16* pw = sP[wave];
    #pragma unroll
    for (int jt = 0; jt < 4; ++jt)
      #pragma unroll
      for (int r = 0; r < 8; ++r)
        pw[(r + hi * 8) * PROW + jt * 16 + m] = (__bf16)sc[jt][r];
    asm volatile("" ::: "memory");  // keep LDS store->load program order
    v16bf aP[2];
    #pragma unroll
    for (int kc = 0; kc < 2; ++kc) {
      const __bf16* pr = pw + m * PROW + kc * 32 + hi * 16;
      aP[kc] = cat8(*(const v8bf*)pr, *(const v8bf*)(pr + 8));
    }

    // ---- O += P * V : 8 WMMAs (V fragments from transposed LDS) ----
    const __bf16* ktb = sKT[buf];
    #pragma unroll
    for (int dt = 0; dt < 4; ++dt) {
      #pragma unroll
      for (int kc = 0; kc < 2; ++kc) {
        const __bf16* vp = ktb + (dt * 16 + m) * KTROW + kc * 32 + hi * 16;
        v16bf bv = cat8(*(const v8bf*)vp, *(const v8bf*)(vp + 8));
        acc[dt] = wmma_bf16(aP[kc], bv, acc[dt]);
      }
    }
    __syncthreads();
  }

  // ---- normalize and write out ----
  #pragma unroll
  for (int r = 0; r < 8; ++r) {
    const float inv = 1.0f / lrow[r];
    const int row = q0 + r + hi * 8;
    float* op = Out + ((size_t)(b * kS + row)) * kModel + h * kD + m;
    #pragma unroll
    for (int dt = 0; dt < 4; ++dt) op[dt * 16] = acc[dt][r] * inv;
  }
}

extern "C" void kernel_launch(void* const* d_in, const int* in_sizes, int n_in,
                              void* d_out, int out_size, void* d_ws, size_t ws_size,
                              hipStream_t stream) {
  (void)in_sizes; (void)n_in; (void)d_ws; (void)ws_size; (void)out_size;
  const float* Q = (const float*)d_in[0];  // K, V ignored (reference uses Q for all)
  float* Out = (float*)d_out;
  dim3 grid(kS / kQPerBlock, kH, kB);      // (16, 16, 4)
  ConvSelfAttention_fa_kernel<<<grid, 256, 0, stream>>>(Q, Out);
}
